// HSTUAttention_30227979829254
// MI455X (gfx1250) — compile-verified, matrix operand order checked by reading, not run on hardware
//
#include <hip/hip_runtime.h>
#include <hip/hip_bf16.h>
#include <math.h>

// HSTU attention for MI455X (gfx1250, wave32, WMMA).
// B=2, N=2048, H=1024, NH=16, HD=64.

typedef __attribute__((ext_vector_type(16))) _Float16 v16h;
typedef __attribute__((ext_vector_type(8)))  _Float16 v8h;
typedef __attribute__((ext_vector_type(4)))  _Float16 v4h;
typedef __attribute__((ext_vector_type(8)))  float    v8f;

#define WMMA16(a, b, c) __builtin_amdgcn_wmma_f32_16x16x32_f16( \
    false, (a), false, (b), (short)0, (c), false, false)

// SiLU via native v_exp_f32 + v_rcp_f32 (no IEEE div sequence).
__device__ __forceinline__ float silu_f(float x) {
    return x * __builtin_amdgcn_rcpf(1.0f + __expf(-x));
}

// Load a v16h fragment per the CDNA5 16-bit A/B layout:
// halves 0..7  = K (or col) khalf..khalf+7
// halves 8..15 = K (or col) khalf+16..khalf+23   (khalf = (lane>>4)*8)
__device__ __forceinline__ v16h load_frag(const _Float16* p) {
    v16h f;
    *((v8h*)&f)       = *(const v8h*)p;
    *(((v8h*)&f) + 1) = *(const v8h*)(p + 16);
    return f;
}

// ---------------------------------------------------------------------------
// Kernel: f32 [K x N] -> f16 transposed [N x K].  Writes are coalesced and
// contiguous; reads are strided but each weight is touched exactly once.
// ---------------------------------------------------------------------------
__global__ void cvt_transpose_f16_kernel(const float* __restrict__ src,
                                         _Float16* __restrict__ dst,
                                         int K, int N) {
    int i = blockIdx.x * blockDim.x + threadIdx.x;   // over N*K, dst-linear
    if (i < N * K) {
        int n = i / K;
        int k = i - n * K;
        dst[i] = (_Float16)src[(size_t)k * N + n];
    }
}

// ---------------------------------------------------------------------------
// WMMA GEMM: out = act(A[MxK] @ W[KxN] + bias), W given pre-transposed f16
// as Wt[N x K].  Block tile 128x64, BK=32. 256 threads = 8 waves arranged
// 4(M) x 2(N); each wave computes a 32x32 tile = 2x2 WMMA frags.
// ---------------------------------------------------------------------------
#define BM 128
#define BN 64
#define BK 32
#define LDA 40  // BK + 8 halves pad (row = 80B, 16B aligned)
#define LDB 40

template <bool A_IS_F32, bool SILU, bool OUT_F16>
__global__ __launch_bounds__(256) void gemm_wmma_kernel(
    const float* __restrict__ A32, const _Float16* __restrict__ A16,
    const _Float16* __restrict__ Wt, const float* __restrict__ bias,
    _Float16* __restrict__ out16, float* __restrict__ out32,
    int M, int N, int K)
{
    __shared__ _Float16 As[BM * LDA];
    __shared__ _Float16 Bs[BN * LDB];

    const int t     = threadIdx.x;
    const int lane  = t & 31;
    const int wid   = t >> 5;
    const int waveM = wid >> 1;   // 0..3
    const int waveN = wid & 1;    // 0..1
    const int row0  = blockIdx.x * BM;
    const int col0  = blockIdx.y * BN;
    const int l16   = lane & 15;
    const int khalf = (lane >> 4) * 8;

    v8f zero = {};
    v8f acc[2][2];
    acc[0][0] = zero; acc[0][1] = zero; acc[1][0] = zero; acc[1][1] = zero;

    for (int k0 = 0; k0 < K; k0 += BK) {
        __syncthreads();
        if (A_IS_F32) {
            // 128x32 f32 tile: float4 loads -> v4h LDS stores (4 per thread).
            #pragma unroll
            for (int i = t; i < BM * (BK / 4); i += 256) {
                int r = i >> 3;
                int c = (i & 7) * 4;
                const float4 v =
                    *(const float4*)&A32[(size_t)(row0 + r) * K + k0 + c];
                v4h h = { (_Float16)v.x, (_Float16)v.y,
                          (_Float16)v.z, (_Float16)v.w };
                *(v4h*)&As[r * LDA + c] = h;
            }
        } else {
            // 128x32 f16 tile: v8h copies (2 per thread).
            #pragma unroll
            for (int i = t; i < BM * (BK / 8); i += 256) {
                int r = i >> 2;
                int c = (i & 3) * 8;
                *(v8h*)&As[r * LDA + c] =
                    *(const v8h*)&A16[(size_t)(row0 + r) * K + k0 + c];
            }
        }
        // 64x32 f16 B tile from pre-transposed Wt[N][K]: one v8h per thread.
        {
            int n = t >> 2;
            int c = (t & 3) * 8;
            *(v8h*)&Bs[n * LDB + c] =
                *(const v8h*)&Wt[(size_t)(col0 + n) * K + k0 + c];
        }
        // Prefetch next Wt tile into GL2 (global_prefetch_b8).
        if (k0 + BK < K)
            __builtin_prefetch(&Wt[(size_t)(col0 + (t >> 2)) * K + k0 + BK], 0, 1);
        __syncthreads();

        v16h afrag[2], bfrag[2];
        #pragma unroll
        for (int am = 0; am < 2; ++am) {
            int r = waveM * 32 + am * 16 + l16;
            afrag[am] = load_frag(&As[r * LDA + khalf]);
        }
        #pragma unroll
        for (int bn = 0; bn < 2; ++bn) {
            int n = waveN * 32 + bn * 16 + l16;
            bfrag[bn] = load_frag(&Bs[n * LDB + khalf]);
        }
        acc[0][0] = WMMA16(afrag[0], bfrag[0], acc[0][0]);
        acc[0][1] = WMMA16(afrag[0], bfrag[1], acc[0][1]);
        acc[1][0] = WMMA16(afrag[1], bfrag[0], acc[1][0]);
        acc[1][1] = WMMA16(afrag[1], bfrag[1], acc[1][1]);
    }

    // Epilogue: C/D layout — lane: col = l16, VGPR r: row = r + (lane>>4)*8.
    #pragma unroll
    for (int am = 0; am < 2; ++am) {
        int rowbase = row0 + waveM * 32 + am * 16 + (lane >> 4) * 8;
        #pragma unroll
        for (int bn = 0; bn < 2; ++bn) {
            int col = col0 + waveN * 32 + bn * 16 + l16;
            float bv = bias[col];
            #pragma unroll
            for (int r = 0; r < 8; ++r) {
                float v = acc[am][bn][r] + bv;
                if (SILU) v = silu_f(v);
                size_t idx = (size_t)(rowbase + r) * N + col;
                if (OUT_F16) out16[idx] = (_Float16)v;
                else         out32[idx] = v;
            }
        }
    }
}

// ---------------------------------------------------------------------------
// Attention: per (b,head) slice, 128-query tile per block, stream 64-key tiles.
// scores = silu(Q K^T)/N * mask  (linear accumulation, no softmax) then S @ V.
// ---------------------------------------------------------------------------
#define AN 2048
#define AH 1024
#define ANH 16
#define AHD 64
#define LQK 72  // 64 + 8 halves pad (144B rows, 16B aligned)

__global__ __launch_bounds__(256) void attn_wmma_kernel(
    const _Float16* __restrict__ Qg, const _Float16* __restrict__ Kg,
    const _Float16* __restrict__ Vg, const float* __restrict__ mask,
    float* __restrict__ Og)
{
    __shared__ _Float16 Qs[128 * LQK];      // 128 queries x 64 d
    __shared__ _Float16 Ks[64 * LQK];       // 64 keys x 64 d
    __shared__ _Float16 Vt[64 * LQK];       // transposed: [d][key]
    __shared__ _Float16 Ps[8 * 16 * LQK];   // per-wave probs: 16 q x 64 key

    const int bh   = blockIdx.y;
    const int b    = bh / ANH;
    const int h    = bh % ANH;
    const int q0   = blockIdx.x * 128;
    const int t    = threadIdx.x;
    const int lane = t & 31;
    const int wid  = t >> 5;
    const int l16  = lane & 15;
    const int khalf = (lane >> 4) * 8;

    const _Float16* Qbase = Qg + (size_t)(b * AN + q0) * AH + h * AHD;
    const float* mrow = mask + (size_t)b * AN * AN;
    const float invN = 1.0f / (float)AN;

    // Stage Q tile: v8h copies (4 per thread).
    #pragma unroll
    for (int i = t; i < 128 * 8; i += 256) {
        int r = i >> 3, d = (i & 7) * 8;
        *(v8h*)&Qs[r * LQK + d] = *(const v8h*)&Qbase[(size_t)r * AH + d];
    }
    __syncthreads();

    // A-fragments from Q are loop-invariant per wave (16 query rows).
    v16h aq0, aq1;
    {
        int r = wid * 16 + l16;
        aq0 = load_frag(&Qs[r * LQK + khalf]);        // d 0..31
        aq1 = load_frag(&Qs[r * LQK + 32 + khalf]);   // d 32..63
    }

    v8f zero = {};
    v8f o[4];
    o[0] = zero; o[1] = zero; o[2] = zero; o[3] = zero;

    _Float16* Pw = &Ps[wid * 16 * LQK];

    for (int m0 = 0; m0 < AN; m0 += 64) {
        __syncthreads();
        const _Float16* Kb = Kg + (size_t)(b * AN + m0) * AH + h * AHD;
        const _Float16* Vb = Vg + (size_t)(b * AN + m0) * AH + h * AHD;
        // Stage K (straight copy) and V (transpose-scatter), v8h global loads.
        #pragma unroll
        for (int i = t; i < 64 * 8; i += 256) {
            int r = i >> 3, d = (i & 7) * 8;
            v8h kv = *(const v8h*)&Kb[(size_t)r * AH + d];
            v8h vv = *(const v8h*)&Vb[(size_t)r * AH + d];
            *(v8h*)&Ks[r * LQK + d] = kv;
            #pragma unroll
            for (int j = 0; j < 8; ++j)
                Vt[(d + j) * LQK + r] = vv[j];
        }
        __syncthreads();

        // Scores: 16 queries x 64 keys per wave, 2 WMMAs per 16x16 tile.
        const int qbg = q0 + wid * 16 + (lane >> 4) * 8;  // query row (global)
        #pragma unroll
        for (int nt = 0; nt < 4; ++nt) {
            int kr = nt * 16 + l16;
            v16h bk0 = load_frag(&Ks[kr * LQK + khalf]);
            v16h bk1 = load_frag(&Ks[kr * LQK + 32 + khalf]);
            v8f s = zero;
            s = WMMA16(aq0, bk0, s);
            s = WMMA16(aq1, bk1, s);

            int kc = m0 + nt * 16 + l16;  // key column (global)
            #pragma unroll
            for (int r = 0; r < 8; ++r) {
                float x = s[r];
                float v = silu_f(x) * invN *
                          mrow[(size_t)(qbg + r) * AN + kc];
                Pw[((lane >> 4) * 8 + r) * LQK + nt * 16 + l16] = (_Float16)v;
            }
        }

        // P @ V: re-layout P (C->A) via per-wave LDS round-trip (in-order DS).
        v16h ap0 = load_frag(&Pw[l16 * LQK + khalf]);        // keys 0..31
        v16h ap1 = load_frag(&Pw[l16 * LQK + 32 + khalf]);   // keys 32..63
        #pragma unroll
        for (int dt = 0; dt < 4; ++dt) {
            int dc = dt * 16 + l16;
            v16h bv0 = load_frag(&Vt[dc * LQK + khalf]);
            v16h bv1 = load_frag(&Vt[dc * LQK + 32 + khalf]);
            o[dt] = WMMA16(ap0, bv0, o[dt]);
            o[dt] = WMMA16(ap1, bv1, o[dt]);
        }
    }

    // Write f32 attention output [B*N, H].
    int qb = b * AN + q0 + wid * 16 + (lane >> 4) * 8;
    #pragma unroll
    for (int dt = 0; dt < 4; ++dt) {
        int col = h * AHD + dt * 16 + l16;
        #pragma unroll
        for (int r = 0; r < 8; ++r)
            Og[(size_t)(qb + r) * AH + col] = o[dt][r];
    }
}

// ---------------------------------------------------------------------------
// LayerNorm over H=1024 + build o_input = [U, A, U*A] as f16 [4096 x 3072].
// ---------------------------------------------------------------------------
__global__ __launch_bounds__(256) void ln_concat_kernel(
    const float* __restrict__ attn, const _Float16* __restrict__ Uh,
    _Float16* __restrict__ oin)
{
    __shared__ float red[256];
    const int row = blockIdx.x;
    const int t = threadIdx.x;
    const float* x = attn + (size_t)row * 1024;

    float vals[4], s = 0.0f, ss = 0.0f;
    #pragma unroll
    for (int i = 0; i < 4; ++i) {
        float v = x[t + i * 256];
        vals[i] = v; s += v; ss += v * v;
    }
    red[t] = s; __syncthreads();
    for (int st = 128; st > 0; st >>= 1) {
        if (t < st) red[t] += red[t + st];
        __syncthreads();
    }
    float mean = red[0] * (1.0f / 1024.0f);
    __syncthreads();
    red[t] = ss; __syncthreads();
    for (int st = 128; st > 0; st >>= 1) {
        if (t < st) red[t] += red[t + st];
        __syncthreads();
    }
    float var  = red[0] * (1.0f / 1024.0f) - mean * mean;
    float rstd = __builtin_amdgcn_rsqf(var + 1e-8f);   // native v_rsq_f32

    _Float16* orow = oin + (size_t)row * 3072;
    const _Float16* urow = Uh + (size_t)row * 1024;
    #pragma unroll
    for (int i = 0; i < 4; ++i) {
        int c = t + i * 256;
        float a = (vals[i] - mean) * rstd;
        float u = (float)urow[c];
        orow[c]        = (_Float16)u;
        orow[1024 + c] = (_Float16)a;
        orow[2048 + c] = (_Float16)(u * a);
    }
}

// ---------------------------------------------------------------------------
// Host-side orchestration.
// ---------------------------------------------------------------------------
extern "C" void kernel_launch(void* const* d_in, const int* in_sizes, int n_in,
                              void* d_out, int out_size, void* d_ws, size_t ws_size,
                              hipStream_t stream) {
    const float* query = (const float*)d_in[0];
    const float* key_  = (const float*)d_in[1];
    const float* value = (const float*)d_in[2];
    const float* mask  = (const float*)d_in[3];
    const float* Wq = (const float*)d_in[4];   const float* bq = (const float*)d_in[5];
    const float* Wk = (const float*)d_in[6];   const float* bk = (const float*)d_in[7];
    const float* Wv = (const float*)d_in[8];   const float* bv = (const float*)d_in[9];
    const float* Wu = (const float*)d_in[10];  const float* bu = (const float*)d_in[11];
    const float* Wo = (const float*)d_in[12];  const float* bo = (const float*)d_in[13];
    float* out = (float*)d_out;

    const int M = 2 * 2048;        // 4096 rows
    const int H = 1024;

    // Workspace carve-out (~90 MB total, deterministic layout).
    char* ws = (char*)d_ws;
    size_t off = 0;
    auto carve = [&](size_t bytes) -> void* {
        void* p = ws + off;
        off += (bytes + 255) & ~(size_t)255;
        return p;
    };
    _Float16* Wq_t = (_Float16*)carve((size_t)H * H * 2);       // [N][K] f16
    _Float16* Wk_t = (_Float16*)carve((size_t)H * H * 2);
    _Float16* Wv_t = (_Float16*)carve((size_t)H * H * 2);
    _Float16* Wu_t = (_Float16*)carve((size_t)H * H * 2);
    _Float16* Wo_t = (_Float16*)carve((size_t)H * 3 * H * 2);   // [1024][3072]
    _Float16* Q_h  = (_Float16*)carve((size_t)M * H * 2);
    _Float16* K_h  = (_Float16*)carve((size_t)M * H * 2);
    _Float16* V_h  = (_Float16*)carve((size_t)M * H * 2);
    _Float16* U_h  = (_Float16*)carve((size_t)M * H * 2);
    float*    At_f = (float*)   carve((size_t)M * H * 4);
    _Float16* Oi_h = (_Float16*)carve((size_t)M * 3 * H * 2);
    (void)ws_size; (void)in_sizes; (void)n_in; (void)out_size;

    // 1) Weights -> f16, transposed to [N][K] for contiguous B staging.
    int nw = H * H;
    cvt_transpose_f16_kernel<<<(nw + 255) / 256, 256, 0, stream>>>(Wq, Wq_t, H, H);
    cvt_transpose_f16_kernel<<<(nw + 255) / 256, 256, 0, stream>>>(Wk, Wk_t, H, H);
    cvt_transpose_f16_kernel<<<(nw + 255) / 256, 256, 0, stream>>>(Wv, Wv_t, H, H);
    cvt_transpose_f16_kernel<<<(nw + 255) / 256, 256, 0, stream>>>(Wu, Wu_t, H, H);
    int nwo = 3 * H * H;
    cvt_transpose_f16_kernel<<<(nwo + 255) / 256, 256, 0, stream>>>(Wo, Wo_t, 3 * H, H);

    // 2) Projections: silu(X @ W + b), f16 outputs.
    dim3 ggrid(M / BM, H / BN);
    gemm_wmma_kernel<true, true, true><<<ggrid, 256, 0, stream>>>(
        query, nullptr, Wq_t, bq, Q_h, nullptr, M, H, H);
    gemm_wmma_kernel<true, true, true><<<ggrid, 256, 0, stream>>>(
        key_,  nullptr, Wk_t, bk, K_h, nullptr, M, H, H);
    gemm_wmma_kernel<true, true, true><<<ggrid, 256, 0, stream>>>(
        value, nullptr, Wv_t, bv, V_h, nullptr, M, H, H);
    gemm_wmma_kernel<true, true, true><<<ggrid, 256, 0, stream>>>(
        value, nullptr, Wu_t, bu, U_h, nullptr, M, H, H);

    // 3) Attention (16 query tiles x 32 (b,head) slices).
    dim3 agrid(AN / 128, 2 * ANH);
    attn_wmma_kernel<<<agrid, 256, 0, stream>>>(Q_h, K_h, V_h, mask, At_f);

    // 4) LayerNorm + concat [U, A, U*A].
    ln_concat_kernel<<<M, 256, 0, stream>>>(At_f, U_h, Oi_h);

    // 5) Final GEMM: o_input[4096x3072] @ Wo[3072x1024] + bo -> f32 out.
    dim3 fgrid(M / BM, H / BN);
    gemm_wmma_kernel<false, false, false><<<fgrid, 256, 0, stream>>>(
        nullptr, Oi_h, Wo_t, bo, nullptr, out, M, H, 3 * H);
}
